// TransformerEncoderBlock_43422119362610
// MI455X (gfx1250) — matrix operand order repaired and checked by reading, not compiled, not run on hardware
//
#include <hip/hip_runtime.h>
#include <hip/hip_bf16.h>
#include <math.h>

typedef __attribute__((ext_vector_type(16))) _Float16 h16;
typedef __attribute__((ext_vector_type(8)))  _Float16 h8;
typedef __attribute__((ext_vector_type(8)))  float    f8;

#define EMBED  768
#define NH     12
#define HD     64
#define HIDDEN 3072
#define BATCH  16
#define SEQ    1024
#define TOK    (BATCH*SEQ)   // 16384 token rows

// ---------------------------------------------------------------------------
// WMMA helpers (CDNA5 wave32, 16x16x32 f16 -> f32)
// ---------------------------------------------------------------------------
__device__ __forceinline__ f8 wmma16x16x32(h16 a, h16 b, f8 c) {
  return __builtin_amdgcn_wmma_f32_16x16x32_f16(false, a, false, b, (short)0, c,
                                                false, false);
}

// A-fragment: 16x32 f16 tile, row-major with leading dim `ld` (halves).
// ISA layout: lane<16 holds row m=lane, K = 0..7 (halves 0..7) and 16..23
// (halves 8..15); lane>=16 holds row m=lane-16, K = 8..15 and 24..31.
__device__ __forceinline__ h16 load_a_frag(const _Float16* t, int ld) {
  const unsigned lane = threadIdx.x & 31u;
  const unsigned m = lane & 15u, g = lane >> 4;
  const _Float16* p = t + (size_t)m * ld + g * 8;
  h8 lo = *(const h8*)(p);
  h8 hi = *(const h8*)(p + 16);
  h16 r;
#pragma unroll
  for (int i = 0; i < 8; ++i) { r[i] = lo[i]; r[i + 8] = hi[i]; }
  return r;
}

// B-fragment from B^T storage: t[n][k], leading dim `ld` (halves).
// ISA layout (32x16 B): lane<16 -> col n=lane, K=0..7 and 16..23; lane>=16 ->
// col n=lane-16, K=8..15 and 24..31 (same k order as the A fragment).
__device__ __forceinline__ h16 load_bt_frag(const _Float16* t, int ld) {
  const unsigned lane = threadIdx.x & 31u;
  const _Float16* p = t + (size_t)(lane & 15u) * ld + (lane >> 4) * 8;
  h8 lo = *(const h8*)(p);
  h8 hi = *(const h8*)(p + 16);
  h16 r;
#pragma unroll
  for (int i = 0; i < 8; ++i) { r[i] = lo[i]; r[i + 8] = hi[i]; }
  return r;
}

// ---------------------------------------------------------------------------
// Async global->LDS copy (CDNA5): 16B per lane, no VGPR round trip.
// vdst = per-lane LDS byte address (low 32 bits of generic pointer),
// vaddr = 64-bit global address, saddr = off.  Tracked by ASYNCcnt.
// ---------------------------------------------------------------------------
__device__ __forceinline__ void async_ld_b128(void* lds, const void* gptr) {
  const unsigned lds_addr = (unsigned)(uintptr_t)lds;   // LDS offset = addr[31:0]
  asm volatile("global_load_async_to_lds_b128 %0, %1, off"
               :: "v"(lds_addr), "v"(gptr) : "memory");
}
__device__ __forceinline__ void wait_async0() {
  asm volatile("s_wait_asynccnt 0x0" ::: "memory");
}

// ---------------------------------------------------------------------------
// Generic tiled WMMA GEMM: C[M,N] = A[M,K] (f16, row-major) * Bt[N,K]^T + bias
// 256 threads = 8 waves (2x4), block tile 128x128, k-tile 32.
// Double-buffered LDS fed by global_load_async_to_lds_b128: next k-tile is
// in flight while the current one feeds WMMA; one s_wait_asynccnt + barrier
// per iteration.
// EP: 0 = scatter q/k/v^T heads, 1 = +resid f32 out, 2 = GELU f16 out,
//     3 = +resid f32 out (fc2).
// ---------------------------------------------------------------------------
#define BM 128
#define BN 128
#define BK 32
#define LDT 40   // padded LDS row (halves) to stagger banks

template<int EP>
__global__ __launch_bounds__(256) void gemm_wmma(
    const _Float16* __restrict__ A, const _Float16* __restrict__ Bt,
    const float* __restrict__ bias, int M, int N, int K,
    float* __restrict__ outf, _Float16* __restrict__ outh,
    const float* __restrict__ resid,
    _Float16* __restrict__ qh, _Float16* __restrict__ kh,
    _Float16* __restrict__ vth)
{
  __shared__ __align__(16) _Float16 sA[2][BM * LDT];
  __shared__ __align__(16) _Float16 sB[2][BN * LDT];
  const int tid  = threadIdx.x;
  const int wave = tid >> 5;
  const int wm   = wave >> 2;   // 0..1  -> 64-row wave tile
  const int wn   = wave & 3;    // 0..3  -> 32-col wave tile
  const int m0   = blockIdx.y * BM;
  const int n0   = blockIdx.x * BN;

  f8 acc[4][2] = {};

  // Issue both 16B chunks of this thread's share of the A and Bt tiles.
  auto issue_tile = [&](int kt, int buf) {
#pragma unroll
    for (int i = 0; i < 2; ++i) {
      const int c = tid + i * 256;           // 0..511 chunk id
      const int r = c >> 2, kc = (c & 3) * 8;
      async_ld_b128(&sA[buf][r * LDT + kc],
                    &A[(size_t)(m0 + r) * K + kt + kc]);
      async_ld_b128(&sB[buf][r * LDT + kc],
                    &Bt[(size_t)(n0 + r) * K + kt + kc]);
    }
  };

  // Prologue: stage first tile.
  issue_tile(0, 0);
  wait_async0();
  __syncthreads();

  int buf = 0;
  for (int kt = 0; kt < K; kt += BK) {
    // Prefetch next k-tile into the other buffer while we compute.
    if (kt + BK < K) issue_tile(kt + BK, buf ^ 1);

    h16 af[4];
    h16 bf[2];
#pragma unroll
    for (int mi = 0; mi < 4; ++mi)
      af[mi] = load_a_frag(&sA[buf][(wm * 64 + mi * 16) * LDT], LDT);
#pragma unroll
    for (int ni = 0; ni < 2; ++ni)
      bf[ni] = load_bt_frag(&sB[buf][(wn * 32 + ni * 16) * LDT], LDT);
#pragma unroll
    for (int mi = 0; mi < 4; ++mi)
#pragma unroll
      for (int ni = 0; ni < 2; ++ni)
        acc[mi][ni] = wmma16x16x32(af[mi], bf[ni], acc[mi][ni]);

    if (kt + BK < K) {
      wait_async0();      // my async writes into buf^1 have landed
      __syncthreads();    // everyone done reading buf, everyone's writes done
      buf ^= 1;
    }
  }

  // Epilogue. C-layout: lane group g=(lane>>4) holds rows g*8+r, col lane&15.
  const int lane = tid & 31;
  const int r0   = (lane >> 4) * 8;
  const int cc   = lane & 15;
#pragma unroll
  for (int mi = 0; mi < 4; ++mi) {
#pragma unroll
    for (int ni = 0; ni < 2; ++ni) {
      const int gn  = n0 + wn * 32 + ni * 16 + cc;
      const float bv = bias[gn];
#pragma unroll
      for (int r = 0; r < 8; ++r) {
        const int gm = m0 + wm * 64 + mi * 16 + r0 + r;
        float v = acc[mi][ni][r] + bv;
        if (EP == 0) {
          // Scatter into q[b,h,n,d], k[b,h,n,d], v^T[b,h,d,n]
          const int which = gn / EMBED;
          const int ccx   = gn - which * EMBED;
          const int hh = ccx >> 6, dd = ccx & 63;
          const int bi = gm >> 10, nn = gm & 1023;
          const size_t bh = (size_t)bi * NH + hh;
          if (which == 0)      qh[(bh * SEQ + nn) * HD + dd] = (_Float16)v;
          else if (which == 1) kh[(bh * SEQ + nn) * HD + dd] = (_Float16)v;
          else                 vth[(bh * HD + dd) * SEQ + nn] = (_Float16)v;
        } else if (EP == 1 || EP == 3) {
          const size_t idx = (size_t)gm * N + gn;
          outf[idx] = v + resid[idx];
        } else {  // EP == 2: exact GELU -> f16
          const float ge = 0.5f * v * (1.0f + erff(v * 0.70710678118654752f));
          outh[(size_t)gm * N + gn] = (_Float16)ge;
        }
      }
    }
  }
}

// ---------------------------------------------------------------------------
// Flash attention: 128 threads = 4 waves; each wave owns 16 query rows.
// Keys processed in 64-wide chunks; online softmax with half-wave shuffles.
// ---------------------------------------------------------------------------
__global__ __launch_bounds__(128) void flash_attn(
    const _Float16* __restrict__ Q, const _Float16* __restrict__ Kmat,
    const _Float16* __restrict__ Vt, _Float16* __restrict__ Oout)
{
  __shared__ __align__(16) _Float16 sP[4][16 * 72];
  const int lane = threadIdx.x & 31;
  const int wave = threadIdx.x >> 5;
  const int bh = blockIdx.y;
  const int b = bh / NH, h = bh % NH;
  const int q0 = blockIdx.x * 64 + wave * 16;

  const _Float16* qbase = Q    + ((size_t)bh * SEQ + q0) * HD;
  const _Float16* kbase = Kmat + (size_t)bh * SEQ * HD;
  const _Float16* vbase = Vt   + (size_t)bh * HD * SEQ;

  // Q fragments (head dim 64 -> 2 k-fragments), kept in registers.
  h16 qa0 = load_a_frag(qbase, HD);
  h16 qa1 = load_a_frag(qbase + 32, HD);

  f8 o[4] = {};                 // 16 rows x 64 dims accumulator
  float mrow[8], lrow[8];
#pragma unroll
  for (int r = 0; r < 8; ++r) { mrow[r] = -3.0e38f; lrow[r] = 0.0f; }

  _Float16* myP = sP[wave];
  const float scale = 0.125f;   // 1/sqrt(64)

  for (int j0 = 0; j0 < SEQ; j0 += 64) {
    // S = Q * K^T for 16 queries x 64 keys (K rows are already Bt layout).
    f8 s[4];
#pragma unroll
    for (int ni = 0; ni < 4; ++ni) {
      const _Float16* kb = kbase + (size_t)(j0 + ni * 16) * HD;
      h16 b0 = load_bt_frag(kb, HD);        // dims 0..31
      h16 b1 = load_bt_frag(kb + 32, HD);   // dims 32..63
      f8 a = {};
      a = wmma16x16x32(qa0, b0, a);
      a = wmma16x16x32(qa1, b1, a);
      s[ni] = a;
    }
    // Online softmax: row stats via half-wave (16-lane) xor-shuffle reductions.
#pragma unroll
    for (int r = 0; r < 8; ++r) {
      float smax = -3.0e38f;
#pragma unroll
      for (int ni = 0; ni < 4; ++ni) {
        const float t = s[ni][r] * scale;
        s[ni][r] = t;
        smax = fmaxf(smax, t);
      }
#pragma unroll
      for (int off = 1; off < 16; off <<= 1)
        smax = fmaxf(smax, __shfl_xor(smax, off, 32));
      const float mnew  = fmaxf(mrow[r], smax);
      const float alpha = __expf(mrow[r] - mnew);
      mrow[r] = mnew;
      float rs = 0.0f;
#pragma unroll
      for (int ni = 0; ni < 4; ++ni) {
        const float pv = __expf(s[ni][r] - mnew);
        s[ni][r] = pv;
        rs += pv;
      }
#pragma unroll
      for (int off = 1; off < 16; off <<= 1)
        rs += __shfl_xor(rs, off, 32);
      lrow[r] = lrow[r] * alpha + rs;
#pragma unroll
      for (int nd = 0; nd < 4; ++nd) o[nd][r] *= alpha;
    }
    // C-layout -> A-layout conversion of P through per-wave LDS tile.
    {
      const int pr = (lane >> 4) * 8, pc = lane & 15;
#pragma unroll
      for (int r = 0; r < 8; ++r)
#pragma unroll
        for (int ni = 0; ni < 4; ++ni)
          myP[(pr + r) * 72 + ni * 16 + pc] = (_Float16)s[ni][r];
    }
    h16 pa0 = load_a_frag(myP, 72);       // keys j0+0..31
    h16 pa1 = load_a_frag(myP + 32, 72);  // keys j0+32..63
    // O += P * V  (V^T[d][keypos] rows are contiguous -> Bt fragments)
#pragma unroll
    for (int nd = 0; nd < 4; ++nd) {
      const _Float16* vb = vbase + (size_t)(nd * 16) * SEQ + j0;
      h16 b0 = load_bt_frag(vb, SEQ);
      h16 b1 = load_bt_frag(vb + 32, SEQ);
      o[nd] = wmma16x16x32(pa0, b0, o[nd]);
      o[nd] = wmma16x16x32(pa1, b1, o[nd]);
    }
  }

  // Normalize and store into [b, n, h*64+d] (f16 rows for proj GEMM).
  const int pr = (lane >> 4) * 8, pc = lane & 15;
#pragma unroll
  for (int r = 0; r < 8; ++r) {
    const float inv = 1.0f / lrow[r];
    const size_t row = (size_t)b * SEQ + q0 + pr + r;
#pragma unroll
    for (int nd = 0; nd < 4; ++nd)
      Oout[row * EMBED + h * HD + nd * 16 + pc] = (_Float16)(o[nd][r] * inv);
  }
}

// ---------------------------------------------------------------------------
// LayerNorm (C=768): one 256-thread block per row, 3 elements per thread.
// Writes f16 activations for the WMMA GEMMs.
// ---------------------------------------------------------------------------
__global__ __launch_bounds__(256) void layernorm_f16(
    const float* __restrict__ x, const float* __restrict__ gw,
    const float* __restrict__ bw, _Float16* __restrict__ out)
{
  __shared__ float rs[256], rss[256];
  const int row = blockIdx.x, tid = threadIdx.x;
  const float* xr = x + (size_t)row * EMBED;
  float v[3];
  float s = 0.0f, ss = 0.0f;
#pragma unroll
  for (int i = 0; i < 3; ++i) {
    v[i] = xr[tid + i * 256];
    s += v[i];
    ss += v[i] * v[i];
  }
  rs[tid] = s; rss[tid] = ss;
  __syncthreads();
  for (int off = 128; off > 0; off >>= 1) {
    if (tid < off) { rs[tid] += rs[tid + off]; rss[tid] += rss[tid + off]; }
    __syncthreads();
  }
  const float mean = rs[0] * (1.0f / 768.0f);
  const float var  = rss[0] * (1.0f / 768.0f) - mean * mean;
  const float rstd = rsqrtf(var + 1e-5f);
  _Float16* orow = out + (size_t)row * EMBED;
#pragma unroll
  for (int i = 0; i < 3; ++i) {
    const int c = tid + i * 256;
    orow[c] = (_Float16)(((v[i] - mean) * rstd) * gw[c] + bw[c]);
  }
}

// ---------------------------------------------------------------------------
// Weight convert + transpose: W[K][N] f32 -> Wt[N][K] f16.
// Grid: (N/256, K); coalesced reads along N, strided f16 writes.
// ---------------------------------------------------------------------------
__global__ __launch_bounds__(256) void conv_transpose_f16(
    const float* __restrict__ w, _Float16* __restrict__ wt, int K, int N)
{
  const int n = blockIdx.x * 256 + threadIdx.x;
  const int k = blockIdx.y;
  wt[(size_t)n * K + k] = (_Float16)w[(size_t)k * N + n];
}

// ---------------------------------------------------------------------------
extern "C" void kernel_launch(void* const* d_in, const int* in_sizes, int n_in,
                              void* d_out, int out_size, void* d_ws,
                              size_t ws_size, hipStream_t stream)
{
  const float* x      = (const float*)d_in[0];
  const float* ln1_g  = (const float*)d_in[1];
  const float* ln1_b  = (const float*)d_in[2];
  const float* qkv_w  = (const float*)d_in[3];
  const float* qkv_b  = (const float*)d_in[4];
  const float* proj_w = (const float*)d_in[5];
  const float* proj_b = (const float*)d_in[6];
  const float* ln2_g  = (const float*)d_in[7];
  const float* ln2_b  = (const float*)d_in[8];
  const float* fc1_w  = (const float*)d_in[9];
  const float* fc1_b  = (const float*)d_in[10];
  const float* fc2_w  = (const float*)d_in[11];
  const float* fc2_b  = (const float*)d_in[12];
  float* out = (float*)d_out;

  // Workspace layout (~190 MB; h buffer aliases dead q/k/v^T/attn region).
  char* p = (char*)d_ws;
  _Float16* lnbuf   = (_Float16*)p; p += (size_t)TOK * EMBED * 2;
  _Float16* qkv_wt  = (_Float16*)p; p += (size_t)3 * EMBED * EMBED * 2;
  _Float16* proj_wt = (_Float16*)p; p += (size_t)EMBED * EMBED * 2;
  _Float16* fc1_wt  = (_Float16*)p; p += (size_t)HIDDEN * EMBED * 2;
  _Float16* fc2_wt  = (_Float16*)p; p += (size_t)EMBED * HIDDEN * 2;
  float*    x1      = (float*)p;    p += (size_t)TOK * EMBED * 4;
  _Float16* qh      = (_Float16*)p; p += (size_t)TOK * EMBED * 2;
  _Float16* kh      = (_Float16*)p; p += (size_t)TOK * EMBED * 2;
  _Float16* vth     = (_Float16*)p; p += (size_t)TOK * EMBED * 2;
  _Float16* attnh   = (_Float16*)p; p += (size_t)TOK * EMBED * 2;
  _Float16* hbuf    = qh;  // [TOK][3072] f16 == 4 x 25.2MB region above

  // 1) Weight conversion to f16, transposed to [N][K] for Bt fragments.
  conv_transpose_f16<<<dim3(3 * EMBED / 256, EMBED),  256, 0, stream>>>(qkv_w,  qkv_wt,  EMBED,  3 * EMBED);
  conv_transpose_f16<<<dim3(EMBED / 256, EMBED),      256, 0, stream>>>(proj_w, proj_wt, EMBED,  EMBED);
  conv_transpose_f16<<<dim3(HIDDEN / 256, EMBED),     256, 0, stream>>>(fc1_w,  fc1_wt,  EMBED,  HIDDEN);
  conv_transpose_f16<<<dim3(EMBED / 256, HIDDEN),     256, 0, stream>>>(fc2_w,  fc2_wt,  HIDDEN, EMBED);

  // 2) LN1 -> f16
  layernorm_f16<<<TOK, 256, 0, stream>>>(x, ln1_g, ln1_b, lnbuf);

  // 3) QKV GEMM [16384x768]x[768x2304], scatter to head layouts
  gemm_wmma<0><<<dim3(3 * EMBED / BN, TOK / BM), 256, 0, stream>>>(
      lnbuf, qkv_wt, qkv_b, TOK, 3 * EMBED, EMBED,
      nullptr, nullptr, nullptr, qh, kh, vth);

  // 4) Flash attention (192 bh pairs x 16 query blocks of 64)
  flash_attn<<<dim3(SEQ / 64, BATCH * NH), 128, 0, stream>>>(qh, kh, vth, attnh);

  // 5) Projection GEMM + residual -> x1 (f32)
  gemm_wmma<1><<<dim3(EMBED / BN, TOK / BM), 256, 0, stream>>>(
      attnh, proj_wt, proj_b, TOK, EMBED, EMBED,
      x1, nullptr, x, nullptr, nullptr, nullptr);

  // 6) LN2 -> f16
  layernorm_f16<<<TOK, 256, 0, stream>>>(x1, ln2_g, ln2_b, lnbuf);

  // 7) FC1 GEMM + GELU -> h (f16)
  gemm_wmma<2><<<dim3(HIDDEN / BN, TOK / BM), 256, 0, stream>>>(
      lnbuf, fc1_wt, fc1_b, TOK, HIDDEN, EMBED,
      nullptr, hbuf, nullptr, nullptr, nullptr, nullptr);

  // 8) FC2 GEMM + residual -> output (f32)
  gemm_wmma<3><<<dim3(EMBED / BN, TOK / BM), 256, 0, stream>>>(
      hbuf, fc2_wt, fc2_b, TOK, EMBED, HIDDEN,
      out, nullptr, x1, nullptr, nullptr, nullptr);
}
